// CausalSelfAttention_32650341384264
// MI455X (gfx1250) — matrix operand order repaired and checked by reading, not compiled, not run on hardware
//
#include <hip/hip_runtime.h>

// ---------------------------------------------------------------------------
// CDNA5 (gfx1250) causal self-attention, f16 WMMA w/ f32 accumulation,
// async global->LDS DMA (ASYNCcnt) double-buffered GEMMs.
// B=4, T=2048, E=1024, H=16, hd=64.  Effective logit scale = 1/sqrt(64).
// ---------------------------------------------------------------------------

typedef __attribute__((ext_vector_type(16))) _Float16 v16h;
typedef __attribute__((ext_vector_type(8)))  _Float16 v8h;
typedef __attribute__((ext_vector_type(4)))  _Float16 v4h;
typedef __attribute__((ext_vector_type(8)))  float    v8f;

// 128-bit payload type + address-space-qualified pointers for the async
// global->LDS builtin (probe: param0 = AS1 'int __vector_size__(16)__ *').
typedef __attribute__((__vector_size__(16))) int vi4;
typedef __attribute__((address_space(1))) vi4 as1_vi4;
typedef __attribute__((address_space(3))) vi4 as3_vi4;

#define B_ 4
#define T_ 2048
#define E_ 1024
#define H_ 16
#define HD_ 64

static __device__ __forceinline__ v16h cat16(v8h a, v8h b) {
  return __builtin_shufflevector(a, b, 0,1,2,3,4,5,6,7,8,9,10,11,12,13,14,15);
}
static __device__ __forceinline__ v8f wmma_f16(v16h a, v16h b, v8f c) {
  return __builtin_amdgcn_wmma_f32_16x16x32_f16(false, a, false, b, (short)0, c,
                                                false, false);
}

// ---- async global->LDS (16B per lane), guarded so the file always builds ---
#if __has_builtin(__builtin_amdgcn_global_load_async_to_lds_b128)
#define HAVE_ASYNC_LDS 1
#else
#define HAVE_ASYNC_LDS 0
#endif

static __device__ __forceinline__ void async_copy16(void* lds_dst,
                                                    const void* gsrc) {
#if HAVE_ASYNC_LDS
  __builtin_amdgcn_global_load_async_to_lds_b128(
      (as1_vi4*)gsrc, (as3_vi4*)lds_dst, 0, 0);
#else
  *(v8h*)lds_dst = *(const v8h*)gsrc;
#endif
}
static __device__ __forceinline__ void async_wait0() {
#if HAVE_ASYNC_LDS
#if __has_builtin(__builtin_amdgcn_s_wait_asynccnt)
  __builtin_amdgcn_s_wait_asynccnt(0);
#else
  asm volatile("s_wait_asynccnt 0x0" ::: "memory");
#endif
#endif
}

// ===========================================================================
// Kernel 0a: x (f32) -> xh (f16), same layout [8192,1024].
// ===========================================================================
__global__ __launch_bounds__(256)
void convert_x(const float* __restrict__ X, _Float16* __restrict__ Xh) {
  const size_t i = ((size_t)blockIdx.x * 256 + threadIdx.x) * 4;
  const float4 f = *(const float4*)(X + i);
  v4h h = {(_Float16)f.x, (_Float16)f.y, (_Float16)f.z, (_Float16)f.w};
  *(v4h*)(Xh + i) = h;
}

// ===========================================================================
// Kernel 0b: W[k][n] (f32) -> Wt[n][k] (f16), 64x64 tiles via LDS transpose.
// blockIdx.z selects Wq/Wk/Wv/Wp; outputs are contiguous in ws.
// ===========================================================================
__global__ __launch_bounds__(256)
void convert_wt(const float* __restrict__ Wq, const float* __restrict__ Wk,
                const float* __restrict__ Wv, const float* __restrict__ Wp,
                _Float16* __restrict__ Wt0) {
  __shared__ __align__(16) _Float16 Tl[64][68];
  const int z = blockIdx.z;
  const float* W = (z == 0) ? Wq : (z == 1) ? Wk : (z == 2) ? Wv : Wp;
  _Float16* Wt = Wt0 + (size_t)z * E_ * E_;
  const int n0 = blockIdx.x * 64, k0 = blockIdx.y * 64;
  const int tid = threadIdx.x;

  #pragma unroll
  for (int p = 0; p < 4; ++p) {       // read W[k0+r][n0+c..c+3], coalesced
    const int r = p * 16 + (tid >> 4);
    const int c = (tid & 15) * 4;
    const float4 f = *(const float4*)(W + (size_t)(k0 + r) * E_ + n0 + c);
    v4h h = {(_Float16)f.x, (_Float16)f.y, (_Float16)f.z, (_Float16)f.w};
    *(v4h*)&Tl[r][c] = h;
  }
  __syncthreads();

  const int n = tid >> 2;             // write Wt[n0+n][k0+kc..kc+15]
  const int kc = (tid & 3) * 16;
  __align__(16) _Float16 tmp[16];
  #pragma unroll
  for (int i = 0; i < 16; ++i) tmp[i] = Tl[kc + i][n];
  _Float16* dst = Wt + (size_t)(n0 + n) * E_ + k0 + kc;
  *(v8h*)dst = *(v8h*)&tmp[0];
  *(v8h*)(dst + 8) = *(v8h*)&tmp[8];
}

// ===========================================================================
// Kernel 1: GEMM  C[8192,1024] = A(f16) @ Wt^T + bias.
// mode = modeBase + blockIdx.z: 0=Q, 1=K, 2=V(transposed out), 3=Y (f32 out).
// 128x128x32 block tile, 8 waves x (32x64), double-buffered async LDS DMA.
// ===========================================================================
__global__ __launch_bounds__(256)
void gemm_f16(const _Float16* __restrict__ Axh, const _Float16* __restrict__ Ao,
              const _Float16* __restrict__ Wt0,
              const float* __restrict__ bq, const float* __restrict__ bk,
              const float* __restrict__ bv, const float* __restrict__ bp,
              _Float16* __restrict__ q, _Float16* __restrict__ k,
              _Float16* __restrict__ vt, float* __restrict__ Y, int modeBase) {
  __shared__ __align__(16) _Float16 As[2][128][40];   // [m][k], pad 40
  __shared__ __align__(16) _Float16 Bs[2][128][40];   // [n][k] (Wt rows)

  const int mode = modeBase + blockIdx.z;
  const _Float16* A = (mode == 3) ? Ao : Axh;
  const _Float16* Wt = Wt0 + (size_t)mode * E_ * E_;
  const float* bias = (mode == 0) ? bq : (mode == 1) ? bk
                    : (mode == 2) ? bv : bp;

  const int tid  = threadIdx.x;
  const int lane = tid & 31, wave = tid >> 5;
  const int hi   = (lane >> 4) & 1, l15 = lane & 15;
  const int kb8  = hi * 8, kb16 = hi * 16;
  const int waveM = wave & 3, waveN = wave >> 2;
  const int blockM = blockIdx.y * 128, blockN = blockIdx.x * 128;

  // copy role: each thread moves 32B of one A row and one Bt row per tile
  const int cr = tid >> 1;            // row 0..127
  const int cb = (tid & 1) * 32;      // byte offset within 64B row

  v8f acc[2][4];
  #pragma unroll
  for (int mt = 0; mt < 2; ++mt)
    #pragma unroll
    for (int nt = 0; nt < 4; ++nt)
      acc[mt][nt] = (v8f){0.f,0.f,0.f,0.f,0.f,0.f,0.f,0.f};

  // issue async copies of the 128x32 f16 tiles for k-step k0 into buffer buf
  #define ISSUE_TILE(buf, k0)                                                  \
    do {                                                                       \
      const char* ga = (const char*)(A + (size_t)(blockM + cr) * E_ + (k0));   \
      const char* gb = (const char*)(Wt + (size_t)(blockN + cr) * E_ + (k0));  \
      char* la = (char*)&As[buf][cr][0];                                       \
      char* lb = (char*)&Bs[buf][cr][0];                                       \
      async_copy16(la + cb,      ga + cb);                                     \
      async_copy16(la + cb + 16, ga + cb + 16);                                \
      async_copy16(lb + cb,      gb + cb);                                     \
      async_copy16(lb + cb + 16, gb + cb + 16);                                \
    } while (0)

  ISSUE_TILE(0, 0);

  for (int it = 0; it < E_ / 32; ++it) {
    const int buf = it & 1;
    async_wait0();                    // tile `it` landed in LDS (this wave)
    __syncthreads();                  // ...and in every wave; prev buf free
    if (it + 1 < E_ / 32) ISSUE_TILE(buf ^ 1, (it + 1) * 32);  // overlap DMA

    v16h afr[2], bfr[4];
    #pragma unroll
    for (int mt = 0; mt < 2; ++mt) {
      const _Float16* ap = &As[buf][waveM * 32 + mt * 16 + l15][kb16];
      afr[mt] = cat16(*(const v8h*)ap, *(const v8h*)(ap + 8));
    }
    #pragma unroll
    for (int nt = 0; nt < 4; ++nt) {
      const _Float16* bp_ = &Bs[buf][waveN * 64 + nt * 16 + l15][kb16];
      bfr[nt] = cat16(*(const v8h*)bp_, *(const v8h*)(bp_ + 8));
    }
    #pragma unroll
    for (int mt = 0; mt < 2; ++mt)
      #pragma unroll
      for (int nt = 0; nt < 4; ++nt)
        acc[mt][nt] = wmma_f16(afr[mt], bfr[nt], acc[mt][nt]);
  }
  #undef ISSUE_TILE

  // ---- epilogue ----
  #pragma unroll
  for (int mt = 0; mt < 2; ++mt) {
    #pragma unroll
    for (int nt = 0; nt < 4; ++nt) {
      const int col = blockN + waveN * 64 + nt * 16 + l15;   // 0..1023
      const float bb = bias[col];
      const int h = col >> 6, d = col & 63;
      const int rbase = blockM + waveM * 32 + mt * 16 + hi * 8;
      #pragma unroll
      for (int i = 0; i < 8; ++i) {
        const int row = rbase + i;        // token row = b*T + t
        const int b = row >> 11, t = row & (T_ - 1);
        const float fv = acc[mt][nt][i] + bb;
        if (mode == 0)
          q[(((size_t)(b * H_ + h) * T_ + t) << 6) + d] = (_Float16)fv;
        else if (mode == 1)
          k[(((size_t)(b * H_ + h) * T_ + t) << 6) + d] = (_Float16)fv;
        else if (mode == 2)
          vt[((size_t)(b * H_ + h) * HD_ + d) * T_ + t] = (_Float16)fv;
        else
          Y[(size_t)row * E_ + col] = fv;
      }
    }
  }
}

// ===========================================================================
// Kernel 2: causal flash attention.  One wave per 16-query tile, 32 keys per
// step, online softmax, P routed through per-wave LDS to become A-fragment.
// Q/K: [B,H,T,64] f16.  Vt: [B,H,64,T] f16.  O: [B,T,E] f16 (heads merged).
// ===========================================================================
__global__ __launch_bounds__(256)
void attn_fwd(const _Float16* __restrict__ Q, const _Float16* __restrict__ K,
              const _Float16* __restrict__ Vt, _Float16* __restrict__ O) {
  __shared__ __align__(16) _Float16 P[8][16][40];

  const int tid  = threadIdx.x;
  const int lane = tid & 31, wave = tid >> 5;
  const int hi   = lane >> 4, l15 = lane & 15;
  const int kb8  = hi * 8, kb16 = hi * 16;

  const int gq = blockIdx.x * 8 + wave;
  const int bh = gq >> 7;
  const int q0 = (gq & 127) << 4;

  const _Float16* Qb = Q  + ((size_t)bh * T_ + q0) * HD_;
  const _Float16* Kb = K  + (size_t)bh * T_ * HD_;
  const _Float16* Vb = Vt + (size_t)bh * HD_ * T_;

  v16h aQ[2];
  {
    const _Float16* qrow = Qb + (size_t)l15 * HD_;
    #pragma unroll
    for (int c = 0; c < 2; ++c) {
      const _Float16* p = qrow + c * 32 + kb16;
      aQ[c] = cat16(*(const v8h*)p, *(const v8h*)(p + 8));
    }
  }

  v8f oacc[4];
  #pragma unroll
  for (int nt = 0; nt < 4; ++nt)
    oacc[nt] = (v8f){0.f,0.f,0.f,0.f,0.f,0.f,0.f,0.f};
  float m[8], l[8];
  #pragma unroll
  for (int i = 0; i < 8; ++i) { m[i] = -3.0e38f; l[i] = 0.f; }

  const int nsteps = (q0 + 47) >> 5;
  for (int s = 0; s < nsteps; ++s) {
    const int j = s * 32;
    if (j + 32 < T_)
      __builtin_prefetch(Kb + (size_t)(j + 32 + l15) * HD_, 0, 1);

    v8f s1 = (v8f){0.f,0.f,0.f,0.f,0.f,0.f,0.f,0.f};
    v8f s2 = s1;
    #pragma unroll
    for (int kh = 0; kh < 2; ++kh) {
      const _Float16* krow = Kb + (size_t)(j + kh * 16 + l15) * HD_;
      #pragma unroll
      for (int c = 0; c < 2; ++c) {
        const _Float16* p = krow + c * 32 + kb16;
        v16h bK = cat16(*(const v8h*)p, *(const v8h*)(p + 8));
        if (kh == 0) s1 = wmma_f16(aQ[c], bK, s1);
        else         s2 = wmma_f16(aQ[c], bK, s2);
      }
    }

    const int c1 = j + l15, c2 = j + 16 + l15;
    #pragma unroll
    for (int i = 0; i < 8; ++i) {
      const int row = q0 + i + hi * 8;
      s1[i] = (c1 <= row) ? s1[i] * 0.125f : -3.0e38f;
      s2[i] = (c2 <= row) ? s2[i] * 0.125f : -3.0e38f;
    }

    float corr[8];
    #pragma unroll
    for (int i = 0; i < 8; ++i) {
      float t = fmaxf(s1[i], s2[i]);
      t = fmaxf(t, __shfl_xor(t, 1, 16));
      t = fmaxf(t, __shfl_xor(t, 2, 16));
      t = fmaxf(t, __shfl_xor(t, 4, 16));
      t = fmaxf(t, __shfl_xor(t, 8, 16));
      const float mn = fmaxf(m[i], t);
      corr[i] = __expf(m[i] - mn);
      m[i] = mn;
      const float p1 = __expf(s1[i] - mn);
      const float p2 = __expf(s2[i] - mn);
      s1[i] = p1; s2[i] = p2;
      float rs = p1 + p2;
      rs += __shfl_xor(rs, 1, 16);
      rs += __shfl_xor(rs, 2, 16);
      rs += __shfl_xor(rs, 4, 16);
      rs += __shfl_xor(rs, 8, 16);
      l[i] = l[i] * corr[i] + rs;
    }
    #pragma unroll
    for (int nt = 0; nt < 4; ++nt)
      #pragma unroll
      for (int i = 0; i < 8; ++i)
        oacc[nt][i] *= corr[i];

    #pragma unroll
    for (int i = 0; i < 8; ++i) {
      const int r = i + hi * 8;
      P[wave][r][l15]      = (_Float16)s1[i];
      P[wave][r][16 + l15] = (_Float16)s2[i];
    }
    asm volatile("s_wait_dscnt 0" ::: "memory");
    const _Float16* pr = &P[wave][l15][kb16];
    const v16h aP = cat16(*(const v8h*)pr, *(const v8h*)(pr + 8));

    #pragma unroll
    for (int nt = 0; nt < 4; ++nt) {
      const _Float16* vp = Vb + (size_t)(nt * 16 + l15) * T_ + j + kb16;
      const v16h bV = cat16(*(const v8h*)vp, *(const v8h*)(vp + 8));
      oacc[nt] = wmma_f16(aP, bV, oacc[nt]);
    }
  }

  float inv[8];
  #pragma unroll
  for (int i = 0; i < 8; ++i) inv[i] = 1.0f / l[i];
  const int b = bh >> 4, h = bh & 15;
  _Float16* ob = O + (size_t)(b * T_ + q0) * E_ + h * HD_;
  #pragma unroll
  for (int nt = 0; nt < 4; ++nt)
    #pragma unroll
    for (int i = 0; i < 8; ++i)
      ob[(size_t)(i + hi * 8) * E_ + nt * 16 + l15] =
          (_Float16)(oacc[nt][i] * inv[i]);
}

// ===========================================================================
extern "C" void kernel_launch(void* const* d_in, const int* in_sizes, int n_in,
                              void* d_out, int out_size, void* d_ws,
                              size_t ws_size, hipStream_t stream) {
  const float* x  = (const float*)d_in[0];
  // d_in[1] = ch_list scalar (unused)
  const float* Wq = (const float*)d_in[2];
  const float* bq = (const float*)d_in[3];
  const float* Wk = (const float*)d_in[4];
  const float* bk = (const float*)d_in[5];
  const float* Wv = (const float*)d_in[6];
  const float* bv = (const float*)d_in[7];
  const float* Wp = (const float*)d_in[8];
  const float* bp = (const float*)d_in[9];
  float* out = (float*)d_out;

  const size_t XE  = (size_t)B_ * T_ * E_;   // 8,388,608
  const size_t WE  = (size_t)E_ * E_;        // 1,048,576
  _Float16* xh  = (_Float16*)d_ws;           // [8192,1024] f16
  _Float16* wt  = xh + XE;                   // 4x Wt[n][k] f16 (Q,K,V,P)
  _Float16* qws = wt + 4 * WE;               // [B,H,T,64]
  _Float16* kws = qws + XE;                  // [B,H,T,64]
  _Float16* vws = kws + XE;                  // [B,H,64,T]
  _Float16* ows = xh;                        // O aliases xh (disjoint lifetime)

  // 0) one-time f32->f16 conversions (weights also transposed)
  convert_x<<<dim3(XE / (256 * 4)), 256, 0, stream>>>(x, xh);
  convert_wt<<<dim3(16, 16, 4), 256, 0, stream>>>(Wq, Wk, Wv, Wp, wt);
  // 1) QKV projections (blockIdx.z = 0/1/2)
  gemm_f16<<<dim3(E_ / 128, (B_ * T_) / 128, 3), 256, 0, stream>>>(
      xh, ows, wt, bq, bk, bv, bp, qws, kws, vws, out, 0);
  // 2) causal flash attention
  attn_fwd<<<dim3((B_ * H_ * T_ / 16) / 8), 256, 0, stream>>>(qws, kws, vws,
                                                              ows);
  // 3) output projection (mode 3)
  gemm_f16<<<dim3(E_ / 128, (B_ * T_) / 128, 1), 256, 0, stream>>>(
      xh, ows, wt, bq, bk, bv, bp, qws, kws, vws, out, 3);
}
